// Attention_29892972380716
// MI455X (gfx1250) — matrix operand order repaired and checked by reading, not compiled, not run on hardware
//
#include <hip/hip_runtime.h>
#include <hip/hip_bf16.h>

// ---------------------------------------------------------------------------
// Attention forward for MI455X (gfx1250): bf16 WMMA (f32 accum) with
// Tensor Data Mover (tensor_load_to_lds / s_wait_tensorcnt) tile staging.
//   d_out = [ out(B,N,C) | q(B,H,N,HD) | k | v ]  all f32.
// ---------------------------------------------------------------------------

typedef __attribute__((ext_vector_type(16))) __bf16 v16bf;
typedef __attribute__((ext_vector_type(8)))  float  v8f;
typedef __attribute__((ext_vector_type(2)))  float  v2f;
typedef __attribute__((ext_vector_type(4)))  unsigned v4u;
typedef __attribute__((ext_vector_type(8)))  unsigned v8u;

#define NB 2
#define NN 4096
#define NC 512
#define NH 8
#define HDIM 64
#define SCALE 0.125f              // 64^-0.5
#define OUT_ELEMS (NB*NN*NC)      // floats per output section
#define QKV_ELEMS (NB*NH*NN*HDIM)

union AFrag {
    v16bf v;
    __bf16 e[16];
    unsigned u[8];
};

static __device__ inline v8f wmma_bf16(v16bf a, v16bf b, v8f c) {
    return __builtin_amdgcn_wmma_f32_16x16x32_bf16(false, a, false, b,
                                                   (short)0, c, false, false);
}

// A-fragment pair offset: pair i covers K = base..base+1,
// base = half*16 + 2i   (contiguous order, matches B fills)
static __device__ inline int a_pair_off(int i, int half) {
    return half * 16 + 2 * i;
}

static __device__ inline float halfmax(float x) {
    x = fmaxf(x, __shfl_xor(x, 1, 32));
    x = fmaxf(x, __shfl_xor(x, 2, 32));
    x = fmaxf(x, __shfl_xor(x, 4, 32));
    x = fmaxf(x, __shfl_xor(x, 8, 32));
    return x;
}
static __device__ inline float halfsum(float x) {
    x += __shfl_xor(x, 1, 32);
    x += __shfl_xor(x, 2, 32);
    x += __shfl_xor(x, 4, 32);
    x += __shfl_xor(x, 8, 32);
    return x;
}

// ---- CDNA5 Tensor Data Mover --------------------------------------------
// low 32 bits of a generic pointer into LDS == wave-relative LDS byte address
static __device__ inline unsigned lds_of(const void* p) {
    return (unsigned)(unsigned long long)p;
}

// 2D tensor tile load via TDM. dims/strides/tiles in data_size units.
// ds_code: 0=1B,1=2B,2=4B,3=8B. pad_int: 8B units before pad = 1<<pad_int.
// pad_amt: pad DWORDs-1. One instruction per issuing wave (EXEC ignored).
static __device__ inline void tdm_load_2d(unsigned lds_addr, const void* gsrc,
                                          unsigned ds_code,
                                          unsigned dim0, unsigned dim1,
                                          unsigned stride0,
                                          unsigned tile0, unsigned tile1,
                                          unsigned pad_int, unsigned pad_amt) {
    unsigned long long ga = (unsigned long long)gsrc;
    v4u g0;
    g0[0] = 1u;                                   // count=1, user mode
    g0[1] = lds_addr;                             // lds_addr [63:32]
    g0[2] = (unsigned)ga;                         // global_addr lo
    g0[3] = ((unsigned)(ga >> 32) & 0x01FFFFFFu)  // global_addr [56:32]
            | (2u << 30);                         // type=2 (image)
    v8u g1;
    g1[0] = (ds_code << 16) | (1u << 20)          // data_size, pad_enable
            | (pad_int << 22) | (pad_amt << 25);
    g1[1] = (dim0 & 0xFFFFu) << 16;               // tensor_dim0 lo16 (atomic addr 0)
    g1[2] = (dim0 >> 16) | ((dim1 & 0xFFFFu) << 16);
    g1[3] = (dim1 >> 16) | (tile0 << 16);         // tile_dim0
    g1[4] = tile1 & 0xFFFFu;                      // tile_dim1 (tile_dim2=0)
    g1[5] = stride0;                              // tensor_dim0_stride lo32
    g1[6] = 0u;                                   // stride hi / dim1_stride lo
    g1[7] = 0u;
    asm volatile("tensor_load_to_lds %0, %1" :: "s"(g0), "s"(g1) : "memory");
}
static __device__ inline void wait_tensor0() {
    __builtin_amdgcn_s_wait_tensorcnt(0);
}

// ---------------------------------------------------------------------------
// Kernel 1: qkv = x @ W_qkv^T + hat. TDM-staged f32 tiles, bf16 WMMA.
// grid (64, 24): 128-row M tile, 64-col D tile, 8 waves -> 32x32 each.
// Optionally stages bf16 copies: qb=q*SCALE [B,H,N,HD], kb [B,H,N,HD],
// vbT [B,H,HD,N] for the TDM attention kernel.
// ---------------------------------------------------------------------------
__global__ __launch_bounds__(256) void qkv_kernel(
    const float* __restrict__ x, const float* __restrict__ Wq,
    const float* __restrict__ qh, const float* __restrict__ kh,
    const float* __restrict__ vh,
    float* __restrict__ outq, float* __restrict__ outk, float* __restrict__ outv,
    __bf16* __restrict__ qb, __bf16* __restrict__ kbst, __bf16* __restrict__ vbT,
    int do_stage) {

    __shared__ alignas(16) float xsf[2][128][36];   // 128B rows + 16B pad
    __shared__ alignas(16) float wsf[2][64][36];

    const int m0 = blockIdx.x * 128;
    const int d0 = blockIdx.y * 64;
    const int tid = threadIdx.x, wid = tid >> 5, lane = tid & 31;
    const int half = lane >> 4, l15 = lane & 15;
    const int wm = wid & 3, wd = wid >> 2;

    v8f c[2][2] = {};

    auto issue = [&](int it, int buf) {
        int k0 = it * 32;
        // x tile: 128 rows x 32 f32 (row 128B, pad 16B -> int=4, amt=3)
        tdm_load_2d(lds_of(&xsf[buf][0][0]), x + (size_t)m0 * NC + k0,
                    2u, NC, NB * NN, NC, 32u, 128u, 4u, 3u);
        // W tile: 64 rows x 32 f32
        tdm_load_2d(lds_of(&wsf[buf][0][0]), Wq + (size_t)d0 * NC + k0,
                    2u, NC, 3 * NC, NC, 32u, 64u, 4u, 3u);
    };

    if (wid == 0) issue(0, 0);

    for (int it = 0; it < NC / 32; ++it) {
        if (wid == 0) wait_tensor0();
        __syncthreads();
        const int buf = it & 1;
        if (it + 1 < NC / 32 && wid == 0) issue(it + 1, buf ^ 1);

        AFrag a[2], bfr[2];
#pragma unroll
        for (int mi = 0; mi < 2; ++mi) {
            const float* row = &xsf[buf][wm * 32 + mi * 16 + l15][0];
#pragma unroll
            for (int i = 0; i < 8; ++i) {
                v2f f = *(const v2f*)(row + a_pair_off(i, half));
                a[mi].e[2 * i] = (__bf16)f.x;
                a[mi].e[2 * i + 1] = (__bf16)f.y;
            }
        }
#pragma unroll
        for (int di = 0; di < 2; ++di) {
            const float* row = &wsf[buf][wd * 32 + di * 16 + l15][0];
#pragma unroll
            for (int i = 0; i < 8; ++i) {
                v2f f = *(const v2f*)(row + half * 16 + 2 * i);
                bfr[di].e[2 * i] = (__bf16)f.x;
                bfr[di].e[2 * i + 1] = (__bf16)f.y;
            }
        }
#pragma unroll
        for (int mi = 0; mi < 2; ++mi)
#pragma unroll
            for (int di = 0; di < 2; ++di)
                c[mi][di] = wmma_bf16(a[mi].v, bfr[di].v, c[mi][di]);
    }

    // epilogue: t (q/k/v selector) uniform per block (d0 spans 64 in a 512 band)
    const int t = (int)(blockIdx.y >> 3);
    const float* hat = (t == 0) ? qh : (t == 1) ? kh : vh;
    float* outp      = (t == 0) ? outq : (t == 1) ? outk : outv;

#pragma unroll
    for (int mi = 0; mi < 2; ++mi)
#pragma unroll
        for (int di = 0; di < 2; ++di)
#pragma unroll
            for (int r = 0; r < 8; ++r) {
                int m = m0 + wm * 32 + mi * 16 + r + half * 8;
                int d = d0 + wd * 32 + di * 16 + l15;
                int b = m >> 12, n = m & (NN - 1);
                int rem = d & (NC - 1);
                int h = rem >> 6, hd = rem & (HDIM - 1);
                size_t idx = (((size_t)(b * NH + h)) * NN + n) * HDIM + hd;
                float val = c[mi][di][r] + hat[idx];
                outp[idx] = val;
                if (do_stage) {
                    if (t == 0)      qb[idx] = (__bf16)(val * SCALE);
                    else if (t == 1) kbst[idx] = (__bf16)val;
                    else             vbT[(((size_t)(b * NH + h)) * HDIM + hd) * NN + n] = (__bf16)val;
                }
            }
}

// ---------------------------------------------------------------------------
// Kernel 2a: flash attention, bf16 staged inputs, TDM tile staging.
// grid (N/128, B*H); 8 waves x 16 query rows; 32-key tiles double buffered.
// ---------------------------------------------------------------------------
__global__ __launch_bounds__(256) void attn_kernel_tdm(
    const __bf16* __restrict__ qb, const __bf16* __restrict__ kb,
    const __bf16* __restrict__ vbT, float* __restrict__ o_ws) {

    __shared__ alignas(16) __bf16 kTa[2][32][72];   // [key][dim], 128B+16B pad rows
    __shared__ alignas(16) __bf16 vTa[2][64][40];   // [dim][key], 64B+16B pad rows
    __shared__ alignas(16) __bf16 pscr[8][16][40];  // per-wave P relayout

    const int qblk = blockIdx.x;
    const int bh = blockIdx.y;
    const int b = bh >> 3, h = bh & 7;
    const int tid = threadIdx.x, wid = tid >> 5, lane = tid & 31;
    const int half = lane >> 4, l15 = lane & 15;

    const __bf16* qp  = qb  + (size_t)bh * NN * HDIM;
    const __bf16* kp  = kb  + (size_t)bh * NN * HDIM;
    const __bf16* vtp = vbT + (size_t)bh * HDIM * NN;

    // Q fragments (scale folded into qb), pair loads straight from global.
    AFrag qa[2];
    const __bf16* qrowp = qp + (size_t)(qblk * 128 + wid * 16 + l15) * HDIM;
#pragma unroll
    for (int dh = 0; dh < 2; ++dh)
#pragma unroll
        for (int i = 0; i < 8; ++i)
            qa[dh].u[i] = *(const unsigned*)(qrowp + dh * 32 + a_pair_off(i, half));

    v8f o0 = {}, o1 = {}, o2 = {}, o3 = {};
    float mrow[8], lrow[8];
#pragma unroll
    for (int r = 0; r < 8; ++r) { mrow[r] = -1e30f; lrow[r] = 0.f; }

    auto issue = [&](int j, int buf) {
        const int kb0 = j * 32;
        // K tile: 32 keys x 64 bf16 dims (row 128B, pad 16B -> 4/3)
        tdm_load_2d(lds_of(&kTa[buf][0][0]), kp + (size_t)kb0 * HDIM,
                    1u, HDIM, NN, HDIM, 64u, 32u, 4u, 3u);
        // V^T tile: 64 dims x 32 bf16 keys (row 64B, pad 16B -> 3/3)
        tdm_load_2d(lds_of(&vTa[buf][0][0]), vtp + kb0,
                    1u, NN, HDIM, NN, 32u, 64u, 3u, 3u);
    };

    if (wid == 0) issue(0, 0);

    for (int j = 0; j < NN / 32; ++j) {
        if (wid == 0) wait_tensor0();  // tiles for j landed in LDS
        __syncthreads();               // publish; prior compute also done
        const int buf = j & 1;
        if (j + 1 < NN / 32 && wid == 0) issue(j + 1, buf ^ 1);

        // ---- S = (Q*scale) K^T : 16 rows x 32 keys
        v8f s[2];
#pragma unroll
        for (int nt = 0; nt < 2; ++nt) {
            AFrag kb0f, kb1f;
            const __bf16* krow = &kTa[buf][nt * 16 + l15][0];
#pragma unroll
            for (int i = 0; i < 8; ++i) {
                kb0f.u[i] = *(const unsigned*)(krow + half * 16 + 2 * i);       // dims 0..31
                kb1f.u[i] = *(const unsigned*)(krow + 32 + half * 16 + 2 * i);  // dims 32..63
            }
            v8f acc = {};
            acc = wmma_bf16(qa[0].v, kb0f.v, acc);
            acc = wmma_bf16(qa[1].v, kb1f.v, acc);
            s[nt] = acc;
        }

        // ---- online softmax (rows live in 16-lane halves; r -> row r+half*8)
#pragma unroll
        for (int r = 0; r < 8; ++r) {
            float sm = halfmax(fmaxf(s[0][r], s[1][r]));
            float mnew = fmaxf(mrow[r], sm);
            float corr = __expf(mrow[r] - mnew);
            float p0 = __expf(s[0][r] - mnew);
            float p1 = __expf(s[1][r] - mnew);
            lrow[r] = lrow[r] * corr + halfsum(p0 + p1);
            mrow[r] = mnew;
            o0[r] *= corr; o1[r] *= corr; o2[r] *= corr; o3[r] *= corr;
            int rl = r + half * 8;
            pscr[wid][rl][l15] = (__bf16)p0;
            pscr[wid][rl][16 + l15] = (__bf16)p1;
        }

        // ---- relayout P (C-layout) -> A-fragment via wave-private LDS
        AFrag pa;
        const __bf16* prow = &pscr[wid][l15][0];
#pragma unroll
        for (int i = 0; i < 8; ++i)
            pa.u[i] = *(const unsigned*)(prow + a_pair_off(i, half));

        // ---- O += P V   (K = 32 keys, 4 dim tiles of 16)
#pragma unroll
        for (int dt = 0; dt < 4; ++dt) {
            AFrag vb;
            const __bf16* vrow = &vTa[buf][dt * 16 + l15][0];
#pragma unroll
            for (int i = 0; i < 8; ++i)
                vb.u[i] = *(const unsigned*)(vrow + half * 16 + 2 * i);
            if (dt == 0)      o0 = wmma_bf16(pa.v, vb.v, o0);
            else if (dt == 1) o1 = wmma_bf16(pa.v, vb.v, o1);
            else if (dt == 2) o2 = wmma_bf16(pa.v, vb.v, o2);
            else              o3 = wmma_bf16(pa.v, vb.v, o3);
        }
    }

    const int nbase = qblk * 128 + wid * 16 + half * 8;
#pragma unroll
    for (int r = 0; r < 8; ++r) {
        float inv = 1.0f / lrow[r];
        size_t base = ((size_t)b * NN + nbase + r) * NC + h * HDIM;
        o_ws[base + 0 * 16 + l15] = o0[r] * inv;
        o_ws[base + 1 * 16 + l15] = o1[r] * inv;
        o_ws[base + 2 * 16 + l15] = o2[r] * inv;
        o_ws[base + 3 * 16 + l15] = o3[r] * inv;
    }
}

// ---------------------------------------------------------------------------
// Kernel 2b: fallback flash attention reading f32 q/k/v (small workspace).
// ---------------------------------------------------------------------------
__global__ __launch_bounds__(256) void attn_kernel_f32(
    const float* __restrict__ q, const float* __restrict__ k,
    const float* __restrict__ v, float* __restrict__ o_ws) {

    __shared__ alignas(16) __bf16 kT[2][32][72];
    __shared__ alignas(16) __bf16 vT[2][64][40];
    __shared__ alignas(16) __bf16 pscr[8][16][40];

    const int qblk = blockIdx.x;
    const int bh = blockIdx.y;
    const int b = bh >> 3, h = bh & 7;
    const int tid = threadIdx.x, wid = tid >> 5, lane = tid & 31;
    const int half = lane >> 4, l15 = lane & 15;

    const float* qp = q + (size_t)bh * NN * HDIM;
    const float* kp = k + (size_t)bh * NN * HDIM;
    const float* vp = v + (size_t)bh * NN * HDIM;

    AFrag qa[2];
    const int qrow = qblk * 128 + wid * 16 + l15;
#pragma unroll
    for (int dh = 0; dh < 2; ++dh)
#pragma unroll
        for (int e = 0; e < 16; ++e) {
            int vv = e >> 1, p = e & 1;
            int kk = half * 16 + e + dh * 32;
            qa[dh].e[e] = (__bf16)(qp[(size_t)qrow * HDIM + kk] * SCALE);
        }

    v8f o0 = {}, o1 = {}, o2 = {}, o3 = {};
    float mrow[8], lrow[8];
#pragma unroll
    for (int r = 0; r < 8; ++r) { mrow[r] = -1e30f; lrow[r] = 0.f; }

    auto load_tiles = [&](int j, int buf) {
        const int kb = j * 32;
        for (int idx = tid; idx < 32 * 64; idx += 256) {
            int row = idx >> 6, col = idx & 63;
            kT[buf][row][col] = (__bf16)kp[(size_t)(kb + row) * HDIM + col];
            vT[buf][col][row] = (__bf16)vp[(size_t)(kb + row) * HDIM + col];
        }
    };
    load_tiles(0, 0);

    for (int j = 0; j < NN / 32; ++j) {
        __syncthreads();
        const int buf = j & 1;
        if (j + 1 < NN / 32) load_tiles(j + 1, buf ^ 1);

        v8f s[2];
#pragma unroll
        for (int nt = 0; nt < 2; ++nt) {
            AFrag kb0, kb1;
            const __bf16* krow = &kT[buf][nt * 16 + l15][0];
#pragma unroll
            for (int i = 0; i < 8; ++i) {
                kb0.u[i] = *(const unsigned*)(krow + half * 16 + 2 * i);
                kb1.u[i] = *(const unsigned*)(krow + 32 + half * 16 + 2 * i);
            }
            v8f acc = {};
            acc = wmma_bf16(qa[0].v, kb0.v, acc);
            acc = wmma_bf16(qa[1].v, kb1.v, acc);
            s[nt] = acc;
        }

#pragma unroll
        for (int r = 0; r < 8; ++r) {
            float sm = halfmax(fmaxf(s[0][r], s[1][r]));
            float mnew = fmaxf(mrow[r], sm);
            float corr = __expf(mrow[r] - mnew);
            float p0 = __expf(s[0][r] - mnew);
            float p1 = __expf(s[1][r] - mnew);
            lrow[r] = lrow[r] * corr + halfsum(p0 + p1);
            mrow[r] = mnew;
            o0[r] *= corr; o1[r] *= corr; o2[r] *= corr; o3[r] *= corr;
            int rl = r + half * 8;
            pscr[wid][rl][l15] = (__bf16)p0;
            pscr[wid][rl][16 + l15] = (__bf16)p1;
        }

        AFrag pa;
        const __bf16* prow = &pscr[wid][l15][0];
#pragma unroll
        for (int i = 0; i < 8; ++i)
            pa.u[i] = *(const unsigned*)(prow + a_pair_off(i, half));

#pragma unroll
        for (int dt = 0; dt < 4; ++dt) {
            AFrag vb;
            const __bf16* vrow = &vT[buf][dt * 16 + l15][0];
#pragma unroll
            for (int i = 0; i < 8; ++i)
                vb.u[i] = *(const unsigned*)(vrow + half * 16 + 2 * i);
            if (dt == 0)      o0 = wmma_bf16(pa.v, vb.v, o0);
            else if (dt == 1) o1 = wmma_bf16(pa.v, vb.v, o1);
            else if (dt == 2) o2 = wmma_bf16(pa.v, vb.v, o2);
            else              o3 = wmma_bf16(pa.v, vb.v, o3);
        }
    }

    const int nbase = qblk * 128 + wid * 16 + half * 8;
#pragma unroll
    for (int r = 0; r < 8; ++r) {
        float inv = 1.0f / lrow[r];
        size_t base = ((size_t)b * NN + nbase + r) * NC + h * HDIM;
        o_ws[base + 0 * 16 + l15] = o0[r] * inv;
        o_ws[base + 1 * 16 + l15] = o1[r] * inv;
        o_ws[base + 2 * 16 + l15] = o2[r] * inv;
        o_ws[base + 3 * 16 + l15] = o3[r] * inv;
    }
}

// ---------------------------------------------------------------------------
// Kernel 3: out = o_ws @ W_proj^T + b_proj. TDM-staged f32 tiles. grid (64, 8).
// ---------------------------------------------------------------------------
__global__ __launch_bounds__(256) void proj_kernel(
    const float* __restrict__ o_ws, const float* __restrict__ Wp,
    const float* __restrict__ bias, float* __restrict__ out) {

    __shared__ alignas(16) float xsf[2][128][36];
    __shared__ alignas(16) float wsf[2][64][36];

    const int m0 = blockIdx.x * 128;
    const int d0 = blockIdx.y * 64;
    const int tid = threadIdx.x, wid = tid >> 5, lane = tid & 31;
    const int half = lane >> 4, l15 = lane & 15;
    const int wm = wid & 3, wd = wid >> 2;

    v8f c[2][2] = {};

    auto issue = [&](int it, int buf) {
        int k0 = it * 32;
        tdm_load_2d(lds_of(&xsf[buf][0][0]), o_ws + (size_t)m0 * NC + k0,
                    2u, NC, NB * NN, NC, 32u, 128u, 4u, 3u);
        tdm_load_2d(lds_of(&wsf[buf][0][0]), Wp + (size_t)d0 * NC + k0,
                    2u, NC, NC, NC, 32u, 64u, 4u, 3u);
    };

    if (wid == 0) issue(0, 0);

    for (int it = 0; it < NC / 32; ++it) {
        if (wid == 0) wait_tensor0();
        __syncthreads();
        const int buf = it & 1;
        if (it + 1 < NC / 32 && wid == 0) issue(it + 1, buf ^ 1);

        AFrag a[2], bfr[2];
#pragma unroll
        for (int mi = 0; mi < 2; ++mi) {
            const float* row = &xsf[buf][wm * 32 + mi * 16 + l15][0];
#pragma unroll
            for (int i = 0; i < 8; ++i) {
                v2f f = *(const v2f*)(row + a_pair_off(i, half));
                a[mi].e[2 * i] = (__bf16)f.x;
                a[mi].e[2 * i + 1] = (__bf16)f.y;
            }
        }
#pragma unroll
        for (int di = 0; di < 2; ++di) {
            const float* row = &wsf[buf][wd * 32 + di * 16 + l15][0];
#pragma unroll
            for (int i = 0; i < 8; ++i) {
                v2f f = *(const v2f*)(row + half * 16 + 2 * i);
                bfr[di].e[2 * i] = (__bf16)f.x;
                bfr[di].e[2 * i + 1] = (__bf16)f.y;
            }
        }
#pragma unroll
        for (int mi = 0; mi < 2; ++mi)
#pragma unroll
            for (int di = 0; di < 2; ++di)
                c[mi][di] = wmma_bf16(a[mi].v, bfr[di].v, c[mi][di]);
    }

#pragma unroll
    for (int mi = 0; mi < 2; ++mi)
#pragma unroll
        for (int di = 0; di < 2; ++di)
#pragma unroll
            for (int r = 0; r < 8; ++r) {
                int m = m0 + wm * 32 + mi * 16 + r + half * 8;
                int d = d0 + wd * 32 + di * 16 + l15;
                out[(size_t)m * NC + d] = c[mi][di][r] + bias[d];
            }
}

// ---------------------------------------------------------------------------
extern "C" void kernel_launch(void* const* d_in, const int* in_sizes, int n_in,
                              void* d_out, int out_size, void* d_ws, size_t ws_size,
                              hipStream_t stream) {
    const float* x  = (const float*)d_in[0];
    const float* qh = (const float*)d_in[1];
    const float* kh = (const float*)d_in[2];
    const float* vh = (const float*)d_in[3];
    const float* Wq = (const float*)d_in[4];
    const float* Wp = (const float*)d_in[5];
    const float* bp = (const float*)d_in[6];

    float* out  = (float*)d_out;            // [B,N,C]
    float* outq = out + (size_t)OUT_ELEMS;  // [B,H,N,HD]
    float* outk = out + (size_t)OUT_ELEMS * 2;
    float* outv = out + (size_t)OUT_ELEMS * 3;

    // workspace: o_ws f32 (16MB) | qb bf16 (8MB) | kb bf16 (8MB) | vbT bf16 (8MB)
    const size_t OWS_BYTES = (size_t)OUT_ELEMS * 4;
    const size_t STG_BYTES = (size_t)QKV_ELEMS * 2;
    const size_t NEED = OWS_BYTES + 3 * STG_BYTES;

    float*  o_ws = (float*)d_ws;
    __bf16* qb   = (__bf16*)((char*)d_ws + OWS_BYTES);
    __bf16* kb   = (__bf16*)((char*)d_ws + OWS_BYTES + STG_BYTES);
    __bf16* vbT  = (__bf16*)((char*)d_ws + OWS_BYTES + 2 * STG_BYTES);

    const int do_stage = (ws_size >= NEED) ? 1 : 0;

    qkv_kernel<<<dim3(NB * NN / 128, 3 * NC / 64), 256, 0, stream>>>(
        x, Wq, qh, kh, vh, outq, outk, outv,
        do_stage ? qb : nullptr, do_stage ? kb : nullptr, do_stage ? vbT : nullptr,
        do_stage);

    if (do_stage) {
        attn_kernel_tdm<<<dim3(NN / 128, NB * NH), 256, 0, stream>>>(
            qb, kb, vbT, o_ws);
    } else {
        attn_kernel_f32<<<dim3(NN / 128, NB * NH), 256, 0, stream>>>(
            outq, outk, outv, o_ws);
    }

    proj_kernel<<<dim3(NB * NN / 128, NC / 64), 256, 0, stream>>>(
        o_ws, Wp, bp, out);
}
